// CrossAttention_6536940224544
// MI455X (gfx1250) — matrix operand order repaired and checked by reading, not compiled, not run on hardware
//
#include <hip/hip_runtime.h>

// ---------------------------------------------------------------------------
// LoRA cross-attention for MI455X (gfx1250), wave32, WMMA f32_16x16x32_f16,
// with async global->LDS staging (ASYNCcnt) when the toolchain exposes it.
//
//  k_prep : Weff = W + scaling * B@A   (folds LoRA; f32 -> f16 in ws)
//  k_kv   : K/V = cond @ WeffK/V^T + b (WMMA), zero-padded [B,96,64] f16 in ws
//  k_attn : fused Q-proj -> 8-head attention -> O-proj, all through LDS.
//
// ws layout (bytes):  wq 8K | wk 96K | wv 96K | wo 8K | kpad 192K | vpad 192K
// (total ~0.58 MB; intermediates Q / attn-out never touch HBM)
// ---------------------------------------------------------------------------

typedef _Float16 h16;
typedef __attribute__((ext_vector_type(16))) _Float16 v16h;
typedef __attribute__((ext_vector_type(8)))  _Float16 v8h;
typedef __attribute__((ext_vector_type(8)))  float    v8f;

#define BATCH   16
#define CLAT    64
#define CCOND   768
#define NHEADS  8
#define DH      8
#define SEQ     77
#define LPAD    96
#define HW      4096
#define RANK    8
#define LORA_SCALING 1.0f
#define ATTN_SCALE   0.35355339059327373f   // (64/8)^-0.5

// ws byte offsets
#define OFF_WQ   0u
#define OFF_WK   8192u
#define OFF_WV   106496u
#define OFF_WO   204800u
#define OFF_KPAD 212992u
#define OFF_VPAD 409600u

// ---- CDNA5 async global->LDS (guarded: falls back to VGPR path) -----------
#if defined(__has_builtin)
#if __has_builtin(__builtin_amdgcn_global_load_async_to_lds_b128)
#define ASYNC_LDS 1
#endif
#endif

#ifdef ASYNC_LDS
// builtin signature (from hipcc diagnostic): (global int4*, local int4*, Ii, Ii)
typedef int i32x4 __attribute__((vector_size(16)));
typedef __attribute__((address_space(1))) i32x4 gas_i32x4;
typedef __attribute__((address_space(3))) i32x4 las_i32x4;

static __device__ __forceinline__ void async_g2l_b128(const void* gp, void* lp) {
  gas_i32x4* g1 = (gas_i32x4*)(unsigned long long)(size_t)gp;
  las_i32x4* l3 = (las_i32x4*)(unsigned int)(size_t)lp;
  __builtin_amdgcn_global_load_async_to_lds_b128(g1, l3, 0, 0);
}
static __device__ __forceinline__ void wait_async0() {
#if __has_builtin(__builtin_amdgcn_s_wait_asynccnt)
  __builtin_amdgcn_s_wait_asynccnt(0);
#else
  asm volatile("s_wait_asynccnt 0" ::: "memory");
#endif
}
#endif

// 16-bit A-operand K index held by half h of lane-group g (ISA 7.12.2):
// lanes 0-15: VGPR0-3 -> K0..7, VGPR4-7 -> K16..23; lanes 16-31: +8.
static __device__ __forceinline__ int kpat(int h, int g) {
  return (h & 7) + ((h >> 3) << 4) + (g << 3);
}

static __device__ __forceinline__ v8f wmma_f16(v16h a, v16h b, v8f c) {
  return __builtin_amdgcn_wmma_f32_16x16x32_f16(false, a, false, b,
                                                (short)0, c, false, false);
}

static __device__ __forceinline__ v8f zero8() {
  v8f z;
#pragma unroll
  for (int i = 0; i < 8; ++i) z[i] = 0.0f;
  return z;
}

static __device__ __forceinline__ v16h cat8(v8h lo, v8h hi) {
  return __builtin_shufflevector(lo, hi, 0,1,2,3,4,5,6,7,8,9,10,11,12,13,14,15);
}

// ---------------------------------------------------------------------------
// Kernel 1: fold LoRA into effective weights, convert to f16.
// ---------------------------------------------------------------------------
__global__ __launch_bounds__(256)
void k_prep(const float* Wq, const float* Aq, const float* Bq,
            const float* Wk, const float* Ak, const float* Bk,
            const float* Wv, const float* Av, const float* Bv,
            const float* Wo, const float* Ao, const float* Bo,
            h16* wq, h16* wk, h16* wv, h16* wo) {
  const int NQ = CLAT * CLAT, NK = CLAT * CCOND;
  int idx = blockIdx.x * blockDim.x + threadIdx.x;
  const float *W, *A, *Bm; h16* dst; int in_d, off;
  if      (idx < NQ)            { W=Wq; A=Aq; Bm=Bq; dst=wq; in_d=CLAT;  off=idx; }
  else if (idx < NQ + NK)       { W=Wk; A=Ak; Bm=Bk; dst=wk; in_d=CCOND; off=idx-NQ; }
  else if (idx < NQ + 2*NK)     { W=Wv; A=Av; Bm=Bv; dst=wv; in_d=CCOND; off=idx-NQ-NK; }
  else if (idx < 2*NQ + 2*NK)   { W=Wo; A=Ao; Bm=Bo; dst=wo; in_d=CLAT;  off=idx-NQ-2*NK; }
  else return;
  int o = off / in_d, i = off % in_d;
  float acc = W[off];
#pragma unroll
  for (int r = 0; r < RANK; ++r)
    acc += LORA_SCALING * Bm[o * RANK + r] * A[r * in_d + i];
  dst[off] = (h16)acc;
}

// ---------------------------------------------------------------------------
// Kernel 2: K/V projection.  grid = 16 batches * 2 (k|v) * 6 m-tiles.
// cond tile staged ONCE per block in LDS (async on CDNA5), shared by 4 waves.
// Rows >= 77 zero-filled in LDS -> branch-free A fragments, clean padding.
// ---------------------------------------------------------------------------
__global__ __launch_bounds__(128)
void k_kv(const float* cond, const float* bk, const float* bv,
          const h16* wk, const h16* wv, h16* kpad, h16* vpad) {
  int bid = blockIdx.x;
  int b = bid / 12, rr = bid % 12, kv = rr / 6, mtile = rr % 6;
  const h16*  wsrc = kv ? wv : wk;
  const float* bia = kv ? bv : bk;
  h16* dst         = kv ? vpad : kpad;

  int wave = threadIdx.x >> 5, lane = threadIdx.x & 31;
  int g = lane >> 4, ln = lane & 15;
  int ntile = wave;
  int cout = ntile * 16 + ln;          // output channel (B column: N=ln)

  __shared__ __align__(16) float clds[16][772];   // 772*4 = 3088 B (16B mult)

  // stage 16 cond rows (zero past SEQ); 192 x 16B chunks per row
  for (int c = threadIdx.x; c < 16 * 192; c += blockDim.x) {
    int r = c / 192, col4 = (c % 192) * 4;
    int row = mtile * 16 + r;
    if (row < SEQ) {
#ifdef ASYNC_LDS
      async_g2l_b128(cond + ((size_t)b * SEQ + row) * CCOND + col4,
                     &clds[r][col4]);
#else
      const float* cp = cond + ((size_t)b * SEQ + row) * CCOND + col4;
#pragma unroll
      for (int j = 0; j < 4; ++j) clds[r][col4 + j] = cp[j];
#endif
    } else {
#pragma unroll
      for (int j = 0; j < 4; ++j) clds[r][col4 + j] = 0.0f;
    }
  }
#ifdef ASYNC_LDS
  wait_async0();
#endif
  __syncthreads();

  v8f acc = zero8();
  for (int kk = 0; kk < CCOND / 32; ++kk) {
    v16h a;                             // A: M = ln (token row), f32->f16
#pragma unroll
    for (int h = 0; h < 8; ++h) a[h]     = (h16)clds[ln][kk * 32 + 8 * g + h];
#pragma unroll
    for (int h = 0; h < 8; ++h) a[8 + h] = (h16)clds[ln][kk * 32 + 16 + 8 * g + h];
    const h16* wp = wsrc + (size_t)cout * CCOND + kk * 32 + 8 * g;
    v16h bb = cat8(*(const v8h*)wp, *(const v8h*)(wp + 16));
    acc = wmma_f16(a, bb, acc);
  }
  float bval = bia[cout];
#pragma unroll
  for (int r = 0; r < 8; ++r) {
    int m = mtile * 16 + r + 8 * g;               // D: M = r + 8*(lane/16)
    h16 v = (m < SEQ) ? (h16)(acc[r] + bval) : (h16)0.0f;
    dst[((size_t)b * LPAD + m) * CLAT + cout] = v;
  }
}

// ---------------------------------------------------------------------------
// Kernel 3: fused Q-proj + attention + O-proj.
// grid = 16 batches * 64 blocks of 64 query rows; 4 waves / block.
// ---------------------------------------------------------------------------
__global__ __launch_bounds__(128)
void k_attn(const float* zin, const float* bq, const float* bo,
            const h16* wq, const h16* wo,
            const h16* kpad, const h16* vpad, float* out) {
  int b = blockIdx.x >> 6;
  int pblock = blockIdx.x & 63;
  int wave = threadIdx.x >> 5, lane = threadIdx.x & 31;
  int g = lane >> 4, ln = lane & 15;

  // zs (phases 0/1) overlaps pstage (phase 2) in the same LDS bytes.
  __shared__ __align__(16) char smem_u[64 * 68 * 4];   // 17408 >= 13312 (pst)
  __shared__ __align__(16) h16  qlds[64][72];
  __shared__ __align__(16) h16  ao[64][72];
  float (*zs)[68]       = (float (*)[68])smem_u;       // 272 B rows (16B mult)
  h16   (*pst)[16][104] = (h16 (*)[16][104])smem_u;

  // ---- phase 0: stage z tile [C=64][64 rows], coalesced along HW ----
#ifdef ASYNC_LDS
  {
    const float* gz = zin + (size_t)b * CLAT * HW + pblock * 64;
    for (int c = threadIdx.x; c < 1024; c += blockDim.x) {   // 64 rows x 16 chunks
      int ci = c >> 4, pl4 = (c & 15) << 2;
      async_g2l_b128(gz + (size_t)ci * HW + pl4, &zs[ci][pl4]);
    }
    wait_async0();
  }
#else
  for (int idx = threadIdx.x; idx < 64 * 64; idx += blockDim.x) {
    int ci = idx >> 6, pl = idx & 63;
    zs[ci][pl] = zin[((size_t)b * CLAT + ci) * HW + pblock * 64 + pl];
  }
#endif
  // warm L2/WGP$ with this batch's K/V (24 KB) before phase 2
  {
    const char* kp0 = (const char*)(kpad + (size_t)b * LPAD * CLAT);
    const char* vp0 = (const char*)(vpad + (size_t)b * LPAD * CLAT);
    for (int off = threadIdx.x * 128; off < LPAD * CLAT * 2; off += 128 * 128) {
      __builtin_prefetch(kp0 + off, 0, 3);
      __builtin_prefetch(vp0 + off, 0, 3);
    }
  }
  __syncthreads();

  // ---- phase 1: Q = z_flat @ WeffQ^T + bq -> qlds (f16). wave = M tile ----
  {
    int mtile = wave;
    for (int ntile = 0; ntile < 4; ++ntile) {
      v8f acc = zero8();
      for (int kk = 0; kk < 2; ++kk) {
        v16h a;
#pragma unroll
        for (int h = 0; h < 16; ++h)
          a[h] = (h16)zs[kk * 32 + kpat(h, g)][mtile * 16 + ln];
        const h16* wp = wq + (size_t)(ntile * 16 + ln) * CLAT + kk * 32 + 8 * g;
        v16h bb = cat8(*(const v8h*)wp, *(const v8h*)(wp + 16));
        acc = wmma_f16(a, bb, acc);
      }
      float bias = bq[ntile * 16 + ln];
#pragma unroll
      for (int r = 0; r < 8; ++r)
        qlds[mtile * 16 + r + 8 * g][ntile * 16 + ln] = (h16)(acc[r] + bias);
    }
  }
  __syncthreads();

  // ---- phase 2: attention. wave handles heads {wave, wave+4} ----
  for (int hh = 0; hh < 2; ++hh) {
    int head = wave + 4 * hh;
    for (int mt = 0; mt < 4; ++mt) {
      // A = scaled Q tile [16 x 32]; only K<8 valid (dh=8) -> lanes 0-15, h<8
      v16h qa;
#pragma unroll
      for (int h = 0; h < 16; ++h) qa[h] = (h16)0.0f;
      if (g == 0) {
        v8h qv = *(const v8h*)&qlds[mt * 16 + ln][head * DH];
#pragma unroll
        for (int h = 0; h < DH; ++h)
          qa[h] = (h16)(ATTN_SCALE * (float)qv[h]);
      }
      // S = Q K^T over 5 N-tiles of 16 keys (77 -> 80)
      v8f s[5];
#pragma unroll
      for (int t = 0; t < 5; ++t) s[t] = zero8();
      for (int nt = 0; nt < 5; ++nt) {
        int key = nt * 16 + ln;                 // B column N = key
        v16h kb;
#pragma unroll
        for (int h = 0; h < 16; ++h) kb[h] = (h16)0.0f;
        if (g == 0) {
          v8h kvv = *(const v8h*)(kpad + ((size_t)b * LPAD + key) * CLAT + head * DH);
#pragma unroll
          for (int h = 0; h < DH; ++h) kb[h] = kvv[h];
        }
        s[nt] = wmma_f16(qa, kb, s[nt]);
      }
      if (64 + ln >= SEQ) {                     // mask keys 77..79
#pragma unroll
        for (int r = 0; r < 8; ++r) s[4][r] = -1e30f;
      }
      // row-wise softmax: a row's 16 columns live in one 16-lane group,
      // so xor-shuffles 1/2/4/8 reduce exactly within the row (wave32).
      float mrow[8];
#pragma unroll
      for (int r = 0; r < 8; ++r) {
        float m = s[0][r];
#pragma unroll
        for (int t = 1; t < 5; ++t) m = fmaxf(m, s[t][r]);
        m = fmaxf(m, __shfl_xor(m, 1));
        m = fmaxf(m, __shfl_xor(m, 2));
        m = fmaxf(m, __shfl_xor(m, 4));
        m = fmaxf(m, __shfl_xor(m, 8));
        mrow[r] = m;
      }
#pragma unroll
      for (int r = 0; r < 8; ++r) {
        float ssum = 0.0f;
#pragma unroll
        for (int t = 0; t < 5; ++t) {
          s[t][r] = __expf(s[t][r] - mrow[r]);
          ssum += s[t][r];
        }
        ssum += __shfl_xor(ssum, 1);
        ssum += __shfl_xor(ssum, 2);
        ssum += __shfl_xor(ssum, 4);
        ssum += __shfl_xor(ssum, 8);
        float inv = 1.0f / ssum;
#pragma unroll
        for (int t = 0; t < 5; ++t) s[t][r] *= inv;
      }
      // C-layout -> A-layout transpose of P through per-wave LDS scratch
#pragma unroll
      for (int t = 0; t < 5; ++t)
#pragma unroll
        for (int r = 0; r < 8; ++r)
          pst[wave][r + 8 * g][t * 16 + ln] = (h16)s[t][r];
#pragma unroll
      for (int r = 0; r < 8; ++r)               // zero K-pad keys 80..95
        pst[wave][r + 8 * g][80 + ln] = (h16)0.0f;

      // O = P @ V : M=16 rows, N=16 (8 valid), K=96 -> 3 steps
      v8f o = zero8();
      for (int kk = 0; kk < 3; ++kk) {
        const h16* pp = &pst[wave][ln][kk * 32 + 8 * g];
        v16h pa = cat8(*(const v8h*)pp, *(const v8h*)(pp + 16));
        v16h vb;
#pragma unroll
        for (int h = 0; h < 16; ++h) {
          int key2 = kk * 32 + 8 * g + (h & 7) + ((h >> 3) << 4);      // B: K index
          vb[h] = (ln < DH)
                ? vpad[((size_t)b * LPAD + key2) * CLAT + head * DH + ln]
                : (h16)0.0f;
        }
        o = wmma_f16(pa, vb, o);
      }
      if (ln < DH) {
#pragma unroll
        for (int r = 0; r < 8; ++r)
          ao[mt * 16 + r + 8 * g][head * DH + ln] = (h16)o[r];
      }
    }
  }
  __syncthreads();

  // ---- phase 3: out = WeffO x ao^T + bo  (M=c_out, N=p -> coalesced NCHW) --
  {
    int pt = wave;                              // p-subtile = wave
    for (int mt2 = 0; mt2 < 4; ++mt2) {
      v8f acc = zero8();
      for (int kk = 0; kk < 2; ++kk) {
        const h16* wp2 = wo + (size_t)(mt2 * 16 + ln) * CLAT + kk * 32 + 8 * g;
        v16h a = cat8(*(const v8h*)wp2, *(const v8h*)(wp2 + 16));
        const h16* bp = &ao[pt * 16 + ln][kk * 32 + 8 * g];
        v16h bb = cat8(*(const v8h*)bp, *(const v8h*)(bp + 16));
        acc = wmma_f16(a, bb, acc);
      }
#pragma unroll
      for (int r = 0; r < 8; ++r) {
        int cOut = mt2 * 16 + r + 8 * g;
        int p = pblock * 64 + pt * 16 + ln;     // contiguous across lanes
        out[((size_t)b * CLAT + cOut) * HW + p] = acc[r] + bo[cOut];
      }
    }
  }
}

// ---------------------------------------------------------------------------
extern "C" void kernel_launch(void* const* d_in, const int* in_sizes, int n_in,
                              void* d_out, int out_size, void* d_ws, size_t ws_size,
                              hipStream_t stream) {
  (void)in_sizes; (void)n_in; (void)out_size; (void)ws_size;
  const float* z    = (const float*)d_in[0];
  const float* cond = (const float*)d_in[1];
  const float* Wq = (const float*)d_in[2];  const float* bq = (const float*)d_in[3];
  const float* Aq = (const float*)d_in[4];  const float* Bq = (const float*)d_in[5];
  const float* Ak = (const float*)d_in[8];  const float* Bk = (const float*)d_in[9];
  const float* Wk = (const float*)d_in[6];  const float* bk = (const float*)d_in[7];
  const float* Wv = (const float*)d_in[10]; const float* bv = (const float*)d_in[11];
  const float* Av = (const float*)d_in[12]; const float* Bv = (const float*)d_in[13];
  const float* Wo = (const float*)d_in[14]; const float* bo = (const float*)d_in[15];
  const float* Ao = (const float*)d_in[16]; const float* Bo = (const float*)d_in[17];

  char* ws = (char*)d_ws;
  h16* wq   = (h16*)(ws + OFF_WQ);
  h16* wk   = (h16*)(ws + OFF_WK);
  h16* wv   = (h16*)(ws + OFF_WV);
  h16* wo   = (h16*)(ws + OFF_WO);
  h16* kpad = (h16*)(ws + OFF_KPAD);
  h16* vpad = (h16*)(ws + OFF_VPAD);

  int prep_elems = 2 * CLAT * CLAT + 2 * CLAT * CCOND;   // 106496
  k_prep<<<(prep_elems + 255) / 256, 256, 0, stream>>>(
      Wq, Aq, Bq, Wk, Ak, Bk, Wv, Av, Bv, Wo, Ao, Bo, wq, wk, wv, wo);

  k_kv<<<BATCH * 2 * 6, 128, 0, stream>>>(cond, bk, bv, wk, wv, kpad, vpad);

  k_attn<<<BATCH * 64, 128, 0, stream>>>(z, bq, bo, wq, wo, kpad, vpad,
                                         (float*)d_out);
}
